// FusedRecurrentRank1DCModule_81535659148111
// MI455X (gfx1250) — matrix operand order repaired and checked by reading, not compiled, not run on hardware
//
#include <hip/hip_runtime.h>
#include <hip/hip_bf16.h>

typedef __attribute__((ext_vector_type(16))) _Float16 v16h;
typedef __attribute__((ext_vector_type(8)))  _Float16 v8h;
typedef __attribute__((ext_vector_type(4)))  _Float16 v4h;
typedef __attribute__((ext_vector_type(8)))  float    v8f;

#if __has_builtin(__builtin_amdgcn_tensor_load_to_lds)
#define USE_TDM 1
#else
#define USE_TDM 0
#endif

namespace {

constexpr int NB = 2, NT = 4096, NH = 8, DK = 64, DV = 64;
constexpr int CH  = 64;         // chunk length
constexpr int NCH = NT / CH;    // 64 chunks
constexpr float SCALE = 0.125f; // DK^-0.5

__device__ __forceinline__ v8f wmma_f16(v16h a, v16h b, v8f c) {
  return __builtin_amdgcn_wmma_f32_16x16x32_f16(false, a, false, b, (short)0, c,
                                                false, false);
}

// A fragment (16x32 f16): logical A[m][k] = src[(tm*16+m)*64 + kstep*32 + k]
// ISA layout: m = lane&15; VGPR0-3 K = hi8+0..7, VGPR4-7 K = 16+hi8+0..7.
// Both 8-half runs are 16B aligned -> two ds_load_b128.
__device__ __forceinline__ v16h load_a_rm(const _Float16* src, int lane, int tm,
                                          int kstep) {
  const int m   = tm * 16 + (lane & 15);
  const int hi8 = (lane & 16) ? 8 : 0;
  const _Float16* p = src + m * 64 + kstep * 32 + hi8;
  const v8h lo = *(const v8h*)p;
  const v8h hi = *(const v8h*)(p + 16);
  return __builtin_shufflevector(lo, hi, 0, 1, 2, 3, 4, 5, 6, 7, 8, 9, 10, 11,
                                 12, 13, 14, 15);
}

// B fragment (32x16 f16) from TRANSPOSED storage: logical B[k][n] = srcT[n*64+k]
// ISA layout: n = lane&15; VGPR0-3 K = hi8+0..7, VGPR4-7 K = 16+hi8+0..7.
__device__ __forceinline__ v16h load_b_tr(const _Float16* srcT, int lane, int tn,
                                          int kstep) {
  const int n  = tn * 16 + (lane & 15);
  const int k0 = kstep * 32 + ((lane & 16) ? 8 : 0);
  const _Float16* p = srcT + n * 64 + k0;
  const v8h lo = *(const v8h*)p;
  const v8h hi = *(const v8h*)(p + 16);
  return __builtin_shufflevector(lo, hi, 0, 1, 2, 3, 4, 5, 6, 7, 8, 9, 10, 11,
                                 12, 13, 14, 15);
}

#if USE_TDM
// One TDM descriptor pull of a 64x64 f32 tile (row stride NH*DV) into LDS.
__device__ __forceinline__ void tdm_load_v_chunk(const float* gsrc,
                                                 float* ldsDst) {
  typedef __attribute__((ext_vector_type(4))) unsigned int u32x4;
  typedef __attribute__((ext_vector_type(8))) int i32x8;
  typedef __attribute__((ext_vector_type(4))) int i32x4;
  const unsigned long long ga = (unsigned long long)(uintptr_t)gsrc;
  const unsigned int ldsoff =
      (unsigned int)(uintptr_t)(__attribute__((address_space(3))) void*)ldsDst;
  u32x4 g0;
  g0[0] = 1u;                 // count=1 (valid user descriptor)
  g0[1] = ldsoff;             // lds_addr [63:32]
  g0[2] = (unsigned int)ga;   // global_addr [95:64]
  g0[3] = (unsigned int)((ga >> 32) & 0x01FFFFFFu) | (2u << 30);  // addr+type=2
  i32x8 g1;
  g1[0] = 0x20000;            // workgroup_mask=0, data_size=2 (4 bytes)
  g1[1] = (int)(64u << 16);   // tensor_dim0 = 64 (bits 63:48)
  g1[2] = (int)(64u << 16);   // tensor_dim1 = 64 (bits 95:80)
  g1[3] = (int)(64u << 16);   // tile_dim0   = 64 (bits 127:112)
  g1[4] = 64;                 // tile_dim1   = 64
  g1[5] = 512;                // tensor_dim0_stride = NH*DV
  g1[6] = 0;
  g1[7] = 0;
  const i32x4 z4 = {0, 0, 0, 0};
#if __clang_major__ >= 23
  const i32x8 z8 = {0, 0, 0, 0, 0, 0, 0, 0};
  __builtin_amdgcn_tensor_load_to_lds(g0, g1, z4, z4, z8, 0);
#else
  __builtin_amdgcn_tensor_load_to_lds(g0, g1, z4, z4, 0);
#endif
}

__device__ __forceinline__ void tdm_wait() {
#if __has_builtin(__builtin_amdgcn_s_wait_tensorcnt)
  __builtin_amdgcn_s_wait_tensorcnt(0);
#else
  asm volatile("s_wait_tensorcnt 0x0" ::: "memory");
#endif
}
#endif  // USE_TDM

__global__ __launch_bounds__(128, 1)
void gdn_rank1dc_chunked(const float* __restrict__ q, const float* __restrict__ k,
                         const float* __restrict__ v, const float* __restrict__ g,
                         const float* __restrict__ beta,
                         const float* __restrict__ lam_q,
                         const float* __restrict__ lam_k,
                         const float* __restrict__ state0,
                         const float* __restrict__ bias0,
                         float* __restrict__ out) {
  __shared__ alignas(16) float    S0[DK * DV];    // f32 running matrix state
  __shared__ alignas(16) _Float16 S0hT[DV * DK];  // f16 state, TRANSPOSED [v][f]
  __shared__ alignas(16) _Float16 Kc[CH * DK];    // chunk keys, row-major [t][f]
  __shared__ alignas(16) _Float16 Qc[CH * DK];    // scaled queries; later K'^T
  __shared__ alignas(16) float    U[CH * DV];     // v -> RHS -> u, f32 [t][v]
  __shared__ alignas(16) float    Amat[CH * CH];  // strictly-lower system matrix
  __shared__ alignas(16) _Float16 Mmat[CH * CH];  // incl-lower attn matrix [t][r]
  __shared__ alignas(16) _Float16 UhT[DV * CH];   // f16 u, TRANSPOSED [v][r]
  __shared__ float b0[DV];
  __shared__ float gs[CH], gcum[CH], Gam[CH], decC[CH];
  __shared__ float bet[CH], lqs[CH], lks[CH];

  const int tid  = threadIdx.x;
  const int wave = tid >> 5;
  const int lane = tid & 31;
  const int bh = blockIdx.x;
  const int bb = bh / NH, hh = bh % NH;

  for (int i = tid; i < DK * DV; i += 128) S0[i] = state0[(size_t)bh * DK * DV + i];
  if (tid < DV) b0[tid] = bias0[bh * DV + tid];
  __syncthreads();

  for (int ch = 0; ch < NCH; ++ch) {
    const int t0 = ch * CH;

#if USE_TDM
    // DMA the v chunk straight into U while the VALU stages q/k.
    if (wave == 0)
      tdm_load_v_chunk(v + ((size_t)(bb * NT + t0) * NH + hh) * DK, U);
#endif
    // refresh transposed f16 state operand: S0hT[v][f] = S0[f][v]
    for (int i = tid; i < DK * DV; i += 128) {
      const int vc = i >> 6, f = i & 63;
      S0hT[vc * 64 + f] = (_Float16)S0[f * 64 + vc];
    }
    // stage q (pre-scaled) and k as f16, vectorized b128 loads
    for (int i = tid; i < CH * DK / 4; i += 128) {
      const int t = i >> 4, f4 = (i & 15) * 4;
      const size_t off = ((size_t)(bb * NT + t0 + t) * NH + hh) * DK + f4;
      const float4 qv = *(const float4*)(q + off);
      const float4 kv = *(const float4*)(k + off);
      v4h qh = {(_Float16)(qv.x * SCALE), (_Float16)(qv.y * SCALE),
                (_Float16)(qv.z * SCALE), (_Float16)(qv.w * SCALE)};
      v4h kh = {(_Float16)kv.x, (_Float16)kv.y, (_Float16)kv.z, (_Float16)kv.w};
      *(v4h*)(Qc + t * 64 + f4) = qh;
      *(v4h*)(Kc + t * 64 + f4) = kh;
#if !USE_TDM
      *(float4*)(U + t * 64 + f4) = *(const float4*)(v + off);
#endif
    }
    if (tid < CH) {
      const size_t off = (size_t)(bb * NT + t0 + tid) * NH + hh;
      gs[tid]  = g[off];
      bet[tid] = beta[off];
      lqs[tid] = lam_q[off];
      lks[tid] = lam_k[off];
    }
    if (ch + 1 < NCH && tid < CH) {  // prefetch next chunk rows
      const size_t off = ((size_t)(bb * NT + t0 + CH + tid) * NH + hh) * DK;
      __builtin_prefetch(&q[off], 0, 0);
      __builtin_prefetch(&k[off], 0, 0);
    }
#if USE_TDM
    if (wave == 0) tdm_wait();
#endif
    __syncthreads();

    if (tid == 0) {  // inclusive prefix sum of log-decays
      float run = 0.f;
      for (int t = 0; t < CH; ++t) { run += gs[t]; gcum[t] = run; }
    }
    __syncthreads();
    if (tid < CH) {
      Gam[tid]  = __expf(gcum[tid]);                 // Gamma_t
      decC[tid] = __expf(gcum[CH - 1] - gcum[tid]);  // Gamma_C / Gamma_t
    }
    __syncthreads();

    // ---- RHS: U[t][n] = beta_t * (v - Gamma_t * (k_t^T S0 + lk_t b0)) ----
    {
      const int tm = wave;
      for (int tn = 0; tn < 4; ++tn) {
        v8f c = {};
        for (int ks = 0; ks < 2; ++ks)
          c = wmma_f16(load_a_rm(Kc, lane, tm, ks),
                       load_b_tr(S0hT, lane, tn, ks), c);
        const int n  = tn * 16 + (lane & 15);
        const int mb = tm * 16 + ((lane & 16) ? 8 : 0);
#pragma unroll
        for (int j = 0; j < 8; ++j) {
          const int t = mb + j;
          U[t * DV + n] =
              bet[t] * (U[t * DV + n] - Gam[t] * (c[j] + lks[t] * b0[n]));
        }
      }
    }
    // ---- A (strict lower) from K K^T, M (incl. lower) from Q K^T ----
    {
      const int tm = wave;
      for (int tn = 0; tn < 4; ++tn) {
        v8f ckk = {}, cqk = {};
        for (int ks = 0; ks < 2; ++ks) {
          const v16h bkT = load_b_tr(Kc, lane, tn, ks);  // B[f][r] = Kc[r][f]
          ckk = wmma_f16(load_a_rm(Kc, lane, tm, ks), bkT, ckk);
          cqk = wmma_f16(load_a_rm(Qc, lane, tm, ks), bkT, cqk);
        }
        const int r  = tn * 16 + (lane & 15);
        const int mb = tm * 16 + ((lane & 16) ? 8 : 0);
#pragma unroll
        for (int j = 0; j < 8; ++j) {
          const int t = mb + j;
          const float ratio = __expf(gcum[t] - gcum[r]);
          Amat[t * CH + r] =
              (r < t) ? bet[t] * ratio * (ckk[j] + lks[t] * lks[r]) : 0.f;
          Mmat[t * CH + r] =
              (r <= t) ? (_Float16)(ratio * (cqk[j] + lqs[t] * lks[r]))
                       : (_Float16)0.f;
        }
      }
    }
    __syncthreads();

    // ---- forward substitution: U <- (I+A)^-1 U, 4 rows per barrier ----
    for (int t = 0; t < CH; t += 4) {
      if (tid < DV) {
        float a0 = U[(t + 0) * DV + tid];
        float a1 = U[(t + 1) * DV + tid];
        float a2 = U[(t + 2) * DV + tid];
        float a3 = U[(t + 3) * DV + tid];
        for (int r = 0; r < t; ++r) {
          const float ur = U[r * DV + tid];
          a0 -= Amat[(t + 0) * CH + r] * ur;
          a1 -= Amat[(t + 1) * CH + r] * ur;
          a2 -= Amat[(t + 2) * CH + r] * ur;
          a3 -= Amat[(t + 3) * CH + r] * ur;
        }
        a1 -= Amat[(t + 1) * CH + t] * a0;
        a2 -= Amat[(t + 2) * CH + t] * a0 + Amat[(t + 2) * CH + t + 1] * a1;
        a3 -= Amat[(t + 3) * CH + t] * a0 + Amat[(t + 3) * CH + t + 1] * a1 +
              Amat[(t + 3) * CH + t + 2] * a2;
        U[(t + 0) * DV + tid] = a0;
        U[(t + 1) * DV + tid] = a1;
        U[(t + 2) * DV + tid] = a2;
        U[(t + 3) * DV + tid] = a3;
        UhT[tid * CH + t + 0] = (_Float16)a0;
        UhT[tid * CH + t + 1] = (_Float16)a1;
        UhT[tid * CH + t + 2] = (_Float16)a2;
        UhT[tid * CH + t + 3] = (_Float16)a3;
      }
      __syncthreads();
    }

    // ---- O = Gamma_t * (scale q_t^T S0 + lq_t b0) + (M U)_t ----
    {
      const int tm = wave;
      for (int tn = 0; tn < 4; ++tn) {
        v8f c1 = {}, c2 = {};
        for (int ks = 0; ks < 2; ++ks) {
          c1 = wmma_f16(load_a_rm(Qc, lane, tm, ks),
                        load_b_tr(S0hT, lane, tn, ks), c1);
          c2 = wmma_f16(load_a_rm(Mmat, lane, tm, ks),
                        load_b_tr(UhT, lane, tn, ks), c2);
        }
        const int n  = tn * 16 + (lane & 15);
        const int mb = tm * 16 + ((lane & 16) ? 8 : 0);
#pragma unroll
        for (int j = 0; j < 8; ++j) {
          const int t = mb + j;
          out[((size_t)(bb * NT + t0 + t) * NH + hh) * DV + n] =
              Gam[t] * (c1[j] + lqs[t] * b0[n]) + c2[j];
        }
      }
    }
    __syncthreads();  // phase-6 readers of Qc/b0 are done

    // ---- K'^T[f][r] = decC_r * k_r[f] (into Qc's buffer);
    //      b0 <- gC*b0 + sum_r decC_r lk_r u_r ----
    _Float16* const KpT = Qc;  // Qc is dead after the output phase
    for (int i = tid; i < CH * DK; i += 128) {
      const int f = i >> 6, r = i & 63;
      KpT[f * 64 + r] = (_Float16)((float)Kc[r * 64 + f] * decC[r]);
    }
    if (tid < DV) {
      float acc = Gam[CH - 1] * b0[tid];
      for (int r = 0; r < CH; ++r) acc += decC[r] * lks[r] * U[r * DV + tid];
      b0[tid] = acc;
    }
    __syncthreads();

    // ---- S0 <- gC * S0 + K'^T U ----
    {
      const int tm = wave;
      const float gC = Gam[CH - 1];
      for (int tn = 0; tn < 4; ++tn) {
        v8f c = {};
        for (int ks = 0; ks < 2; ++ks)
          c = wmma_f16(load_a_rm(KpT, lane, tm, ks),
                       load_b_tr(UhT, lane, tn, ks), c);
        const int n  = tn * 16 + (lane & 15);
        const int mb = tm * 16 + ((lane & 16) ? 8 : 0);
#pragma unroll
        for (int j = 0; j < 8; ++j) {
          const int f = mb + j;
          S0[f * DV + n] = gC * S0[f * DV + n] + c[j];
        }
      }
    }
    __syncthreads();
  }

  // ---- final state outputs: o ‖ S_f ‖ b_f ----
  constexpr size_t OFF_S = (size_t)NB * NT * NH * DV;
  constexpr size_t OFF_B = OFF_S + (size_t)NB * NH * DK * DV;
  for (int i = tid; i < DK * DV; i += 128)
    out[OFF_S + (size_t)bh * DK * DV + i] = S0[i];
  if (tid < DV) out[OFF_B + bh * DV + tid] = b0[tid];
}

}  // namespace

extern "C" void kernel_launch(void* const* d_in, const int* in_sizes, int n_in,
                              void* d_out, int out_size, void* d_ws, size_t ws_size,
                              hipStream_t stream) {
  (void)in_sizes; (void)n_in; (void)d_ws; (void)ws_size; (void)out_size;
  const float* q    = (const float*)d_in[0];
  const float* k    = (const float*)d_in[1];
  const float* v    = (const float*)d_in[2];
  const float* g    = (const float*)d_in[3];
  const float* beta = (const float*)d_in[4];
  const float* lq   = (const float*)d_in[5];
  const float* lk   = (const float*)d_in[6];
  const float* st   = (const float*)d_in[7];
  const float* bs   = (const float*)d_in[8];
  gdn_rank1dc_chunked<<<dim3(NB * NH), dim3(128), 0, stream>>>(
      q, k, v, g, beta, lq, lk, st, bs, (float*)d_out);
}
